// LocalResidualMatcher_45715631899339
// MI455X (gfx1250) — matrix operand order repaired and checked by reading, not compiled, not run on hardware
//
#include <hip/hip_runtime.h>
#include <math.h>

typedef _Float16 v16h __attribute__((ext_vector_type(16)));
typedef _Float16 v8h  __attribute__((ext_vector_type(8)));
typedef float    v8f  __attribute__((ext_vector_type(8)));
typedef int      v4i  __attribute__((vector_size(16)));   // int4 for async-LDS builtin

#define DEVFN __device__ __forceinline__
#define AS1 __attribute__((address_space(1)))
#define AS3 __attribute__((address_space(3)))

#if defined(__gfx1250__) && __has_builtin(__builtin_amdgcn_global_load_async_to_lds_b128)
#define HAS_ASYNC_LDS 1
#else
#define HAS_ASYNC_LDS 0
#endif

static constexpr int      DIMV   = 64;
static constexpr unsigned VOX    = 64u * 64u * 64u;      // 262144 voxels
static constexpr int      PD     = 66;                   // zero-padded spatial dim
static constexpr unsigned PVOL   = 66u * 66u * 66u;      // 287496
static constexpr int      C1     = 160;                  // 130 channels padded to 5*32
static constexpr int      NK     = 125;                  // 5^3 offsets

DEVFN float ntn_f(float x)  { return isfinite(x) ? x : 0.0f; }
DEVFN float gelu_f(float x) { return 0.5f * x * (1.0f + erff(x * 0.70710678118f)); }

// ---------------------------------------------------------------- zero fill
__global__ void k_zero_h(_Float16* p, unsigned long long n) {
  unsigned long long i  = (unsigned long long)blockIdx.x * blockDim.x + threadIdx.x;
  unsigned long long st = (unsigned long long)gridDim.x * blockDim.x;
  for (; i < n; i += st) p[i] = (_Float16)0.0f;
}
__global__ void k_zero_f(float* p, unsigned long long n) {
  unsigned long long i  = (unsigned long long)blockIdx.x * blockDim.x + threadIdx.x;
  unsigned long long st = (unsigned long long)gridDim.x * blockDim.x;
  for (; i < n; i += st) p[i] = 0.0f;
}

// ------------------------------------------------- weight repack to B-frags
// wB flat layout: [tap][kg][g(4 cout groups)][lane(32)][half(16)]
// B 32x16 f16 layout: lane n<16 holds col n, K=0..7,16..23; lane n+16 holds col n, K=8..15,24..31.
__global__ void k_repack(const float* __restrict__ w, _Float16* __restrict__ wB,
                         int Cin, int taps, int nkg) {
  unsigned long long total = (unsigned long long)taps * nkg * 2048ull;
  unsigned long long i  = (unsigned long long)blockIdx.x * blockDim.x + threadIdx.x;
  unsigned long long st = (unsigned long long)gridDim.x * blockDim.x;
  for (; i < total; i += st) {
    int h  = (int)(i & 15);
    int l  = (int)((i >> 4) & 31);
    int g  = (int)((i >> 9) & 3);
    unsigned long long rest = i >> 11;
    int kg  = (int)(rest % (unsigned)nkg);
    int tap = (int)(rest / (unsigned)nkg);
    int cin  = kg * 32 + ((l < 16) ? 0 : 8) + ((h < 8) ? h : (h + 8));
    int cout = g * 16 + (l & 15);
    float val = (cin < Cin) ? w[((unsigned long long)cout * Cin + cin) * taps + tap] : 0.0f;
    wB[i] = (_Float16)val;
  }
}

// ------------------------------------------- projection (1x1x1) + L2 norm
// GEMM: A = 16 voxels x 32 cin, B = 32 cin x 16 cout (x4 groups -> 64 cout)
__global__ __launch_bounds__(32)
void k_proj(const float* __restrict__ src, const float* __restrict__ bias,
            const _Float16* __restrict__ wB, _Float16* __restrict__ out) {
  __shared__ float lds[16 * 64];
  __shared__ float lsc[16];
  const int l = (int)threadIdx.x;
  const unsigned v0 = blockIdx.x * 16u;
  const unsigned vm = v0 + (unsigned)(l & 15);
  const int kb = (l < 16) ? 0 : 8;

  v16h a;
#pragma unroll
  for (int h = 0; h < 16; ++h) {
    int k = kb + ((h < 8) ? h : (h + 8));
    a[h] = (_Float16)src[(unsigned long long)k * VOX + vm];
  }

  const int rowhi = (l < 16) ? 0 : 8;
#pragma unroll
  for (int g = 0; g < 4; ++g) {
    v16h b = *(const v16h*)(wB + g * 512 + l * 16);
    v8f c;
#pragma unroll
    for (int r = 0; r < 8; ++r) c[r] = 0.0f;
    c = __builtin_amdgcn_wmma_f32_16x16x32_f16(false, a, false, b, (short)0, c, false, false);
    int cn = g * 16 + (l & 15);
    float bv = bias[cn];
#pragma unroll
    for (int r = 0; r < 8; ++r) lds[(r + rowhi) * 64 + cn] = c[r] + bv;
  }
  __syncthreads();
  if (l < 16) {
    float ss = 0.0f;
#pragma unroll 8
    for (int c = 0; c < 64; ++c) { float t = lds[l * 64 + c]; ss += t * t; }
    lsc[l] = 1.0f / fmaxf(sqrtf(ss), 1e-12f);
  }
  __syncthreads();
  for (int i = l; i < 1024; i += 32) {
    int m = i >> 6, c = i & 63;
    out[(unsigned long long)(v0 + m) * 64 + c] = (_Float16)(lds[i] * lsc[m]);
  }
}

// --------------------------- correlation (125 offsets) + softmax statistics
// one voxel per thread; 125 corr values staged in LDS; writes corr + 5 stat
// channels into the padded f16 conv1 input and 6 float output channels.
__global__ __launch_bounds__(64)
void k_corr_stats(const _Float16* __restrict__ sp, const _Float16* __restrict__ tp,
                  _Float16* __restrict__ enc, float* __restrict__ out) {
  __shared__ float ls[64 * NK];
  const int t = (int)threadIdx.x;
  const unsigned v = blockIdx.x * 64u + (unsigned)t;
  const int z = (int)(v >> 12), y = (int)((v >> 6) & 63), x = (int)(v & 63);

  v8h s[8];
  const v8h* sb = (const v8h*)(sp + (unsigned long long)v * 64);
#pragma unroll
  for (int i = 0; i < 8; ++i) s[i] = sb[i];

  float* row = &ls[t * NK];
  const unsigned long long encb =
      (((unsigned long long)(z + 1) * PD + (y + 1)) * PD + (x + 1)) * (unsigned long long)C1;

  float m1 = -3.0e38f, m2 = -3.0e38f;
  for (int k = 0; k < NK; ++k) {
    int dz = k / 25 - 2, dy = (k / 5) % 5 - 2, dx = k % 5 - 2;
    int zc = min(63, max(0, z + dz));
    int yc = min(63, max(0, y + dy));
    int xc = min(63, max(0, x + dx));
    unsigned nv = ((unsigned)zc * 64u + (unsigned)yc) * 64u + (unsigned)xc;
    const v8h* tb = (const v8h*)(tp + (unsigned long long)nv * 64);
    float acc = 0.0f;
#pragma unroll
    for (int i = 0; i < 8; ++i) {
      v8h tv = tb[i];
#pragma unroll
      for (int j = 0; j < 8; ++j) acc += (float)s[i][j] * (float)tv[j];
    }
    row[k] = acc;
    enc[encb + k] = (_Float16)acc;
    if (acc > m1) { m2 = m1; m1 = acc; } else if (acc > m2) { m2 = acc; }
  }

  float Z = 0.0f, sz = 0.0f, sy = 0.0f, sx = 0.0f, se = 0.0f;
  for (int k = 0; k < NK; ++k) {
    float c = row[k];
    float p = expf(c - m1);
    Z += p; se += p * c;
    sz += p * (float)(k / 25 - 2);
    sy += p * (float)((k / 5) % 5 - 2);
    sx += p * (float)(k % 5 - 2);
  }
  float inv = 1.0f / fmaxf(Z, 1e-8f);
  float eoz = ntn_f(sz * inv), eoy = ntn_f(sy * inv), eox = ntn_f(sx * inv);
  float conf = fminf(inv, 1.0f);
  float marg = ntn_f(fmaxf(conf - expf(m2 - m1) * inv, 0.0f));
  float ent  = ntn_f(m1 + logf(fmaxf(Z, 1e-8f)) - se * inv);

  out[0ull * VOX + v] = eoz;  out[1ull * VOX + v] = eoy;  out[2ull * VOX + v] = eox;
  out[3ull * VOX + v] = conf; out[4ull * VOX + v] = marg; out[5ull * VOX + v] = ent;

  enc[encb + 125] = (_Float16)eoz;  enc[encb + 126] = (_Float16)eoy;
  enc[encb + 127] = (_Float16)eox;  enc[encb + 128] = (_Float16)conf;
  enc[encb + 129] = (_Float16)ent;
}

// ----------------------------------------------- 3x3x3 conv as WMMA GEMM
// tile: 16 voxels (along x) x 64 cout. The 3x3 (z,y) x 18-voxel x inC halo is
// staged into LDS with async global->LDS b128 copies (ASYNCcnt path), then
// WMMA A-fragments are built from ds_load_b128. K loop = 27 taps x nkg x 32ch.
// mode 1: raw f16 -> h1_pad interior;  mode 2: raw f32 -> y2 [v][c].
__global__ __launch_bounds__(32)
void k_conv3x3(const _Float16* __restrict__ in, int inC,
               const _Float16* __restrict__ wB, int nkg,
               const float* __restrict__ bias, int mode,
               _Float16* __restrict__ h1, float* __restrict__ y2,
               float* __restrict__ sums, float* __restrict__ sqs) {
  __shared__ __align__(16) _Float16 hal[9 * 18 * C1];   // max 51,840 B (conv1)
  const int l = (int)threadIdx.x;
  const unsigned v0 = blockIdx.x * 16u;
  const unsigned z = v0 >> 12, yy = (v0 >> 6) & 63, x0 = v0 & 63;
  const int m = l & 15;
  const int kb = (l < 16) ? 0 : 8;
  const int rowhi = (l < 16) ? 0 : 8;

  // ---- stage halo rows (each row is contiguous: 18 voxels * inC halves)
  const int rowHalves    = 18 * inC;
  const int chunksPerRow = rowHalves >> 3;          // 16B chunks
  const int totalChunks  = 9 * chunksPerRow;
  for (int i = l; i < totalChunks; i += 32) {
    int row = i / chunksPerRow;
    int cp  = i - row * chunksPerRow;
    int dz  = row / 3, dy = row - (row / 3) * 3;
    unsigned long long gbase =
        (((unsigned long long)(z + dz) * PD + (yy + dy)) * PD + x0) * (unsigned long long)inC
        + (unsigned long long)cp * 8;
    int lbase = row * rowHalves + cp * 8;
#if HAS_ASYNC_LDS
    __builtin_amdgcn_global_load_async_to_lds_b128(
        (AS1 v4i*)(in + gbase), (AS3 v4i*)(&hal[lbase]), 0, 0);
#else
    *(v8h*)(&hal[lbase]) = *(const v8h*)(in + gbase);
#endif
  }
#if HAS_ASYNC_LDS
#if __has_builtin(__builtin_amdgcn_s_wait_asynccnt)
  __builtin_amdgcn_s_wait_asynccnt(0);
#else
  asm volatile("s_wait_asynccnt 0x0" ::: "memory");
#endif
#endif
  __syncthreads();

  v8f acc[4];
#pragma unroll
  for (int g = 0; g < 4; ++g)
#pragma unroll
    for (int r = 0; r < 8; ++r) acc[g][r] = 0.0f;

  for (int tap = 0; tap < 27; ++tap) {
    int dz = tap / 9, dy = (tap / 3) % 3, dx = tap % 3;
    int row = dz * 3 + dy;
    int abase = (row * 18 + m + dx) * inC + kb;
    for (int kg = 0; kg < nkg; ++kg) {
      const _Float16* ap = &hal[abase + kg * 32];
      v8h lo = *(const v8h*)ap;
      v8h hi = *(const v8h*)(ap + 16);
      v16h a = __builtin_shufflevector(lo, hi, 0, 1, 2, 3, 4, 5, 6, 7,
                                       8, 9, 10, 11, 12, 13, 14, 15);
      const _Float16* bp = wB + (unsigned long long)(tap * nkg + kg) * 2048 + l * 16;
#pragma unroll
      for (int g = 0; g < 4; ++g) {
        v16h b = *(const v16h*)(bp + g * 512);
        acc[g] = __builtin_amdgcn_wmma_f32_16x16x32_f16(false, a, false, b, (short)0,
                                                        acc[g], false, false);
      }
    }
  }

#pragma unroll
  for (int g = 0; g < 4; ++g) {
    int cn = g * 16 + (l & 15);
    float bv = bias[cn];
    float s = 0.0f, q = 0.0f;
#pragma unroll
    for (int r = 0; r < 8; ++r) {
      float val = acc[g][r] + bv;
      s += val; q += val * val;
      unsigned mr = (unsigned)(r + rowhi);
      if (mode == 1) {
        unsigned long long pos =
            (((unsigned long long)(z + 1) * PD + (yy + 1)) * PD + (x0 + mr + 1)) * 64ull + cn;
        h1[pos] = (_Float16)val;
      } else {
        y2[(unsigned long long)(v0 + mr) * 64ull + cn] = val;
      }
    }
    atomicAdd(&sums[cn], s);
    atomicAdd(&sqs[cn], q);
  }
}

// --------------------------------- instance-norm + GELU, in place on h1_pad
__global__ void k_norm_gelu_inplace(_Float16* __restrict__ h,
                                    const float* __restrict__ sums,
                                    const float* __restrict__ sqs) {
  const float invV = 1.0f / (float)VOX;
  unsigned long long i  = (unsigned long long)blockIdx.x * blockDim.x + threadIdx.x;
  unsigned long long st = (unsigned long long)gridDim.x * blockDim.x;
  unsigned long long n  = (unsigned long long)VOX * 64ull;
  for (; i < n; i += st) {
    unsigned c = (unsigned)(i & 63);
    unsigned v = (unsigned)(i >> 6);
    float mu  = sums[c] * invV;
    float var = sqs[c] * invV - mu * mu;
    float rstd = rsqrtf(fmaxf(var, 0.0f) + 1e-5f);
    unsigned z = v >> 12, y = (v >> 6) & 63, x = v & 63;
    unsigned long long pos =
        (((unsigned long long)(z + 1) * PD + (y + 1)) * PD + (x + 1)) * 64ull + c;
    float val = (float)h[pos];
    h[pos] = (_Float16)gelu_f((val - mu) * rstd);
  }
}

// ------------------------- instance-norm + GELU + nan_to_num -> NCDHW d_out
__global__ void k_norm_gelu_out(const float* __restrict__ y2,
                                const float* __restrict__ sums,
                                const float* __restrict__ sqs,
                                float* __restrict__ out) {
  const float invV = 1.0f / (float)VOX;
  unsigned long long i  = (unsigned long long)blockIdx.x * blockDim.x + threadIdx.x;
  unsigned long long st = (unsigned long long)gridDim.x * blockDim.x;
  unsigned long long n  = (unsigned long long)VOX * 64ull;
  for (; i < n; i += st) {
    unsigned c = (unsigned)(i & 63);
    unsigned v = (unsigned)(i >> 6);
    float mu  = sums[c] * invV;
    float var = sqs[c] * invV - mu * mu;
    float rstd = rsqrtf(fmaxf(var, 0.0f) + 1e-5f);
    float val = y2[i];
    out[(unsigned long long)c * VOX + v] = ntn_f(gelu_f((val - mu) * rstd));
  }
}

// =========================================================================
extern "C" void kernel_launch(void* const* d_in, const int* in_sizes, int n_in,
                              void* d_out, int out_size, void* d_ws, size_t ws_size,
                              hipStream_t stream) {
  (void)in_sizes; (void)n_in; (void)out_size; (void)ws_size;
  const float* src    = (const float*)d_in[0];
  const float* tgt    = (const float*)d_in[1];
  const float* w_src  = (const float*)d_in[2];
  const float* b_src  = (const float*)d_in[3];
  const float* w_tgt  = (const float*)d_in[4];
  const float* b_tgt  = (const float*)d_in[5];
  const float* w_enc1 = (const float*)d_in[6];
  const float* b_enc1 = (const float*)d_in[7];
  const float* w_enc2 = (const float*)d_in[8];
  const float* b_enc2 = (const float*)d_in[9];
  float* out = (float*)d_out;

  // ---- workspace carve-up (256B aligned)
  char* base = (char*)d_ws;
  size_t off = 0;
  auto take = [&](size_t bytes) -> void* {
    void* p = base + off;
    off = (off + bytes + 255) & ~(size_t)255;
    return p;
  };
  _Float16* wbS   = (_Float16*)take(2048ull * 2);                 // proj src B-frags
  _Float16* wbT   = (_Float16*)take(2048ull * 2);                 // proj tgt B-frags
  _Float16* wb1   = (_Float16*)take(27ull * 5 * 2048 * 2);        // conv1 B-frags
  _Float16* wb2   = (_Float16*)take(27ull * 2 * 2048 * 2);        // conv2 B-frags
  float*    stats = (float*)take(256ull * 4);                     // sums1|sqs1|sums2|sqs2
  _Float16* sp    = (_Float16*)take((size_t)VOX * 64 * 2);        // proj src [v][c]
  _Float16* tp    = (_Float16*)take((size_t)VOX * 64 * 2);        // proj tgt [v][c]
  _Float16* encp  = (_Float16*)take((size_t)PVOL * C1 * 2);       // padded conv1 input
  _Float16* h1p   = (_Float16*)take((size_t)PVOL * 64 * 2);       // padded conv2 input
  float*    y2    = (float*)take((size_t)VOX * 64 * 4);           // conv2 raw output
  float* sums1 = stats + 0, *sqs1 = stats + 64, *sums2 = stats + 128, *sqs2 = stats + 192;

  // ---- 0) zero padded buffers + stat accumulators (every call; deterministic)
  k_zero_h<<<2048, 256, 0, stream>>>(encp, (unsigned long long)PVOL * C1);
  k_zero_h<<<2048, 256, 0, stream>>>(h1p,  (unsigned long long)PVOL * 64);
  k_zero_f<<<1, 256, 0, stream>>>(stats, 256ull);

  // ---- 1) repack weights into WMMA B-fragment layout
  k_repack<<<8,    256, 0, stream>>>(w_src,  wbS, 32,  1, 1);
  k_repack<<<8,    256, 0, stream>>>(w_tgt,  wbT, 32,  1, 1);
  k_repack<<<1080, 256, 0, stream>>>(w_enc1, wb1, 130, 27, 5);
  k_repack<<<432,  256, 0, stream>>>(w_enc2, wb2, 64,  27, 2);

  // ---- 2) WMMA projections + L2 normalize
  k_proj<<<VOX / 16, 32, 0, stream>>>(src, b_src, wbS, sp);
  k_proj<<<VOX / 16, 32, 0, stream>>>(tgt, b_tgt, wbT, tp);

  // ---- 3) correlation + softmax stats (outputs 0..5, builds conv1 input)
  k_corr_stats<<<VOX / 64, 64, 0, stream>>>(sp, tp, encp, out);

  // ---- 4) conv1 (130->64, 3x3x3) via WMMA, then instance-norm + GELU
  k_conv3x3<<<VOX / 16, 32, 0, stream>>>(encp, C1, wb1, 5, b_enc1, 1,
                                         h1p, nullptr, sums1, sqs1);
  k_norm_gelu_inplace<<<8192, 256, 0, stream>>>(h1p, sums1, sqs1);

  // ---- 5) conv2 (64->64, 3x3x3) via WMMA, then instance-norm + GELU + ntn
  k_conv3x3<<<VOX / 16, 32, 0, stream>>>(h1p, 64, wb2, 2, b_enc2, 2,
                                         nullptr, y2, sums2, sqs2);
  k_norm_gelu_out<<<8192, 256, 0, stream>>>(y2, sums2, sqs2, out + 6ull * VOX);
}
